// RegionProposalNetwork_81441169866881
// MI455X (gfx1250) — compile-verified, matrix operand order checked by reading, not run on hardware
//
#include <hip/hip_runtime.h>
#include <hip/hip_bf16.h>
#include <math.h>

// feature switches for the CDNA5-specific data movers (flip if asm rejects)
#define USE_ASYNC 1
#define USE_TDM   1

// ---------------- problem constants (match reference) ----------------
#define BB    2
#define CC    256
#define HH    136
#define WW    200
#define AA    3
#define NPIX  (BB * HH * WW)        // 54400
#define HWA   (HH * WW * AA)        // 81600
#define PRE_N 2000
#define POST_N 1000
#define CAP   4096
#define NEGS  (-1e9f)
#define LOG_CLAMP 4.135166556742356f
#define MIN_SIZE 0.001f

#define HP (HH + 2)
#define WP (WW + 2)

typedef __attribute__((ext_vector_type(16))) __bf16 v16bf;
typedef __attribute__((ext_vector_type(8)))  float  v8f;
typedef __attribute__((ext_vector_type(4)))  unsigned v4u;
typedef __attribute__((ext_vector_type(8)))  unsigned v8u;

// ---------------- helpers ----------------
__device__ __forceinline__ unsigned short f2bf(float f) {
    unsigned u = __float_as_uint(f);
    unsigned r = (u + 0x7FFFu + ((u >> 16) & 1u)) >> 16;   // RNE
    return (unsigned short)r;
}
__device__ __forceinline__ float bf2f(unsigned short h) {
    return __uint_as_float(((unsigned)h) << 16);
}
__device__ __forceinline__ unsigned ordkey(float f) {
    unsigned u = __float_as_uint(f);
    return (u & 0x80000000u) ? ~u : (u | 0x80000000u);     // monotone encoding
}

// async copy: 16B global -> LDS, tracked by ASYNCcnt
__device__ __forceinline__ void async_ld_b128(void* lds, const void* g) {
#if USE_ASYNC
    unsigned off = (unsigned)(size_t)lds;   // LDS byte offset vgpr
    asm volatile("global_load_async_to_lds_b128 %0, %1, off"
                 :: "v"(off), "v"(g) : "memory");
#else
    *reinterpret_cast<uint4*>(lds) = *reinterpret_cast<const uint4*>(g);
#endif
}
__device__ __forceinline__ void wait_async0() {
#if USE_ASYNC
#if __has_builtin(__builtin_amdgcn_s_wait_asynccnt)
    __builtin_amdgcn_s_wait_asynccnt(0);
#else
    asm volatile("s_wait_asynccnt 0x0" ::: "memory");
#endif
#endif
}
__device__ __forceinline__ void wait_tensor0() {
#if USE_TDM
#if __has_builtin(__builtin_amdgcn_s_wait_tensorcnt)
    __builtin_amdgcn_s_wait_tensorcnt(0);
#else
    asm volatile("s_wait_tensorcnt 0x0" ::: "memory");
#endif
#endif
}

// TDM: stage a [tile_dim1=128 rows x tile_dim0=32 elem] bf16 2D tile into LDS.
// D# per ISA 8.3/8.4: group0 {count=1, lds_addr, global_addr, type=2},
// group1 {data_size=2B, tensor_dim0=256, tensor_dim1=256, tile 32x128, stride0=256}.
__device__ __forceinline__ void tdm_load_tile(void* lds, const void* g) {
#if USE_TDM
    unsigned long long ga = (unsigned long long)(size_t)g;
    v4u g0;
    g0[0] = 1u;                                        // count=1, user mode
    g0[1] = (unsigned)(size_t)lds;                     // lds_addr
    g0[2] = (unsigned)(ga & 0xFFFFFFFFull);            // global_addr[31:0]
    g0[3] = (unsigned)((ga >> 32) & 0x01FFFFFFull) | (2u << 30);  // addr[56:32] | type=2
    v8u g1;
    g1[0] = 0x00010000u;                               // wg_mask=0, data_size=1 (2 bytes)
    g1[1] = (256u << 16);                              // tensor_dim0 = 256 (bits 79:48)
    g1[2] = (256u << 16);                              // tensor_dim1 = 256 (bits 111:80)
    g1[3] = (32u << 16);                               // tile_dim0  = 32  (bits 127:112)
    g1[4] = 128u;                                      // tile_dim1  = 128 (bits 143:128)
    g1[5] = 256u;                                      // tensor_dim0_stride = 256
    g1[6] = 0u;
    g1[7] = 0u;
    asm volatile("tensor_load_to_lds %0, %1" :: "s"(g0), "s"(g1) : "memory");
#else
    (void)lds; (void)g;
#endif
}

// 16 bf16 per lane, 16-bit A/B 16x32 fragment layout:
// lane<16 -> K {kb..kb+7, 16+kb..+7}, kb = (lane>>4)*8
__device__ __forceinline__ v16bf lds_frag16(const unsigned short* p, int kbase) {
    union { v16bf v; uint4 q[2]; } u;
    u.q[0] = *reinterpret_cast<const uint4*>(p + kbase);
    u.q[1] = *reinterpret_cast<const uint4*>(p + 16 + kbase);
    return u.v;
}

// ---------------- K0: pack conv weights to bf16 [tap][co][ci] ----------------
__global__ void k_pack_w(const float* __restrict__ conv_w, unsigned short* __restrict__ wp) {
    int e = blockIdx.x * blockDim.x + threadIdx.x;
    if (e >= 9 * CC * CC) return;
    int ci  = e % CC;
    int t   = e / CC;
    int co  = t % CC;
    int tap = t / CC;
    wp[e] = f2bf(conv_w[(co * CC + ci) * 9 + tap]);
}

// ---------------- K1: features NCHW f32 -> padded NHWC bf16 ----------------
__global__ void k_pack_f(const float* __restrict__ feat, unsigned short* __restrict__ fp) {
    int e = blockIdx.x * blockDim.x + threadIdx.x;
    if (e >= BB * HP * WP * CC) return;
    int c  = e % CC;
    int t  = e / CC;
    int xp = t % WP; t /= WP;
    int yp = t % HP;
    int b  = t / HP;
    float v = 0.f;
    if (yp >= 1 && yp <= HH && xp >= 1 && xp <= WW)
        v = feat[((b * CC + c) * HH + (yp - 1)) * WW + (xp - 1)];
    fp[e] = f2bf(v);
}

// ---------------- K2: 3x3 conv as implicit GEMM, LDS-staged WMMA ----------
// block = 256 threads = 8 waves; block tile 64 pixels x 128 out-channels.
// Per 32-K chunk: wave0 issues one TDM tile load for the 8KB weight tile,
// all threads async-copy the 4KB activation tile, drain ASYNC/TENSORcnt,
// barrier, then each wave runs 4x v_wmma_f32_16x16x32_bf16 from LDS.
__global__ __launch_bounds__(256)
void k_conv_wmma(const unsigned short* __restrict__ featp,
                 const unsigned short* __restrict__ wp,
                 const float* __restrict__ conv_b,
                 unsigned short* __restrict__ fbuf) {
    __shared__ unsigned short sA[64 * 32];    // 4 KB activation tile
    __shared__ unsigned short sB[128 * 32];   // 8 KB weight tile

    const int NT = CC / 128;                  // 2 channel tiles
    int tile = blockIdx.x;
    int tn = tile % NT;
    int tm = tile / NT;

    int tid  = threadIdx.x;
    int lane = tid & 31;
    int wid  = tid >> 5;
    int waveM = wid & 3;
    int waveN = wid >> 2;
    int nidx  = lane & 15;
    int kbase = (lane >> 4) * 8;

    // staging coordinates: thread copies 16B of row (tid>>2), chunk (tid&3)
    int srow = tm * 64 + (tid >> 2);
    int sm = srow < NPIX ? srow : (NPIX - 1);
    int sb  = sm / (HH * WW);
    int srm = sm % (HH * WW);
    int sy = srm / WW, sx = srm % WW;
    int schunk = (tid & 3) * 8;               // halves offset within 32-K row

    v8f acc[4];
    #pragma unroll
    for (int i = 0; i < 4; ++i) acc[i] = (v8f){0,0,0,0,0,0,0,0};

    #pragma unroll
    for (int tap = 0; tap < 9; ++tap) {
        int ky = tap / 3, kx = tap % 3;
        const unsigned short* arow =
            featp + ((((sb * HP) + (sy + ky)) * WP + (sx + kx)) * CC);
        const unsigned short* wtap = wp + tap * CC * CC + tn * 128 * CC;

        for (int kc = 0; kc < CC; kc += 32) {
            // ---- stage B tile via Tensor Data Mover (wave 0 only) ----
            if (wid == 0) {
                tdm_load_tile(&sB[0], wtap + kc);
                wait_tensor0();
            }
            // ---- stage A tile via async global->LDS copies ----
            async_ld_b128(&sA[(tid >> 2) * 32 + schunk], arow + kc + schunk);
            wait_async0();
            __syncthreads();

            // ---- WMMA from LDS ----
            const unsigned short* ap = &sA[(waveM * 16 + nidx) * 32];
            v16bf a = lds_frag16(ap, kbase);
            #pragma unroll
            for (int wn = 0; wn < 4; ++wn) {
                const unsigned short* bp = &sB[(waveN * 64 + wn * 16 + nidx) * 32];
                v16bf bf = lds_frag16(bp, kbase);
                acc[wn] = __builtin_amdgcn_wmma_f32_16x16x32_bf16(
                    false, a, false, bf, (short)0, acc[wn], false, false);
            }
            __syncthreads();   // protect LDS tiles before next stage
        }
    }

    // epilogue: bias + relu, store bf16 NHWC activation for the heads.
    // C/D layout: VGPR r -> lanes 0-15 M=r, lanes 16-31 M=8+r; N = lane&15.
    int colBase = tn * 128 + waveN * 64;
    int mBase = tm * 64 + waveM * 16 + ((lane >= 16) ? 8 : 0);
    #pragma unroll
    for (int wn = 0; wn < 4; ++wn) {
        int col = colBase + wn * 16 + nidx;
        float bias = conv_b[col];
        #pragma unroll
        for (int r = 0; r < 8; ++r) {
            int mr = mBase + r;
            if (mr < NPIX) {
                float v = acc[wn][r] + bias;
                v = fmaxf(v, 0.f);
                fbuf[mr * CC + col] = f2bf(v);
            }
        }
    }
}

// ---------------- K3: heads (256->15) + sigmoid + decode + clip + min-size ----
__global__ void k_heads(const unsigned short* __restrict__ fbuf,
                        const float* __restrict__ cls_w, const float* __restrict__ cls_b,
                        const float* __restrict__ bbox_w, const float* __restrict__ bbox_b,
                        const float* __restrict__ anchors,
                        float img_w, float img_h,
                        float* __restrict__ scores, float* __restrict__ props) {
    int m = blockIdx.x * blockDim.x + threadIdx.x;
    if (m >= NPIX) return;
    int b   = m / (HH * WW);
    int rem = m % (HH * WW);
    int y = rem / WW, x = rem % WW;

    const unsigned short* frow = fbuf + (size_t)m * CC;
    float ac[3]  = {0.f, 0.f, 0.f};
    float ab[12] = {0.f};
    for (int c = 0; c < CC; ++c) {
        float fv = bf2f(frow[c]);
        #pragma unroll
        for (int a = 0; a < 3; ++a) ac[a] = fmaf(fv, cls_w[a * CC + c], ac[a]);
        #pragma unroll
        for (int k = 0; k < 12; ++k) ab[k] = fmaf(fv, bbox_w[k * CC + c], ab[k]);
    }

    int arBase = (y * WW + x) * AA;
    #pragma unroll
    for (int a = 0; a < 3; ++a) {
        float logit = ac[a] + cls_b[a];
        float score = 1.f / (1.f + expf(-logit));
        float dx = ab[a * 4 + 0] + bbox_b[a * 4 + 0];
        float dy = ab[a * 4 + 1] + bbox_b[a * 4 + 1];
        float dw = fminf(ab[a * 4 + 2] + bbox_b[a * 4 + 2], LOG_CLAMP);
        float dh = fminf(ab[a * 4 + 3] + bbox_b[a * 4 + 3], LOG_CLAMP);
        int ar = arBase + a;
        float ax1 = anchors[ar * 4 + 0], ay1 = anchors[ar * 4 + 1];
        float ax2 = anchors[ar * 4 + 2], ay2 = anchors[ar * 4 + 3];
        float wa = ax2 - ax1, ha = ay2 - ay1;
        float cxa = ax1 + 0.5f * wa, cya = ay1 + 0.5f * ha;
        float cx = dx * wa + cxa, cy = dy * ha + cya;
        float w  = expf(dw) * wa, h = expf(dh) * ha;
        float x1 = fminf(fmaxf(cx - 0.5f * w, 0.f), img_w);
        float y1 = fminf(fmaxf(cy - 0.5f * h, 0.f), img_h);
        float x2 = fminf(fmaxf(cx + 0.5f * w, 0.f), img_w);
        float y2 = fminf(fmaxf(cy + 0.5f * h, 0.f), img_h);
        bool valid = ((x2 - x1) >= MIN_SIZE) && ((y2 - y1) >= MIN_SIZE);
        int si = b * HWA + ar;
        scores[si] = valid ? score : NEGS;
        props[si * 4 + 0] = x1; props[si * 4 + 1] = y1;
        props[si * 4 + 2] = x2; props[si * 4 + 3] = y2;
    }
}

// ---------------- K4: top-2000 per batch via radix histogram + bitonic ----
__global__ void k_hist(const float* __restrict__ scores, unsigned* __restrict__ hist) {
    int i = blockIdx.x * blockDim.x + threadIdx.x;
    if (i >= BB * HWA) return;
    int b = i / HWA;
    unsigned bin = ordkey(scores[i]) >> 20;          // 4096 bins
    atomicAdd(&hist[b * 4096 + bin], 1u);
}

__global__ void k_thresh(const unsigned* __restrict__ hist, unsigned* __restrict__ thr) {
    int b = blockIdx.x;
    if (threadIdx.x != 0) return;
    unsigned cum = 0, T = 0;
    for (int bin = 4095; bin >= 0; --bin) {
        cum += hist[b * 4096 + bin];
        if (cum >= PRE_N) { T = (unsigned)bin; break; }
    }
    thr[b] = T;
}

__global__ void k_compact(const float* __restrict__ scores,
                          const unsigned* __restrict__ thr,
                          unsigned* __restrict__ cnt,
                          int* __restrict__ cidx, float* __restrict__ cscore) {
    int i = blockIdx.x * blockDim.x + threadIdx.x;
    if (i >= BB * HWA) return;
    int b = i / HWA;
    float s = scores[i];
    if ((ordkey(s) >> 20) >= thr[b]) {
        unsigned pos = atomicAdd(&cnt[b], 1u);
        if (pos < CAP) {
            cidx[b * CAP + pos]   = i % HWA;
            cscore[b * CAP + pos] = s;
        }
    }
}

__global__ __launch_bounds__(1024)
void k_sort_gather(const unsigned* __restrict__ cnt,
                   const int* __restrict__ cidx, const float* __restrict__ cscore,
                   const float* __restrict__ props,
                   float* __restrict__ tscore, float* __restrict__ tbox) {
    __shared__ float ls[CAP];
    __shared__ int   li[CAP];
    int b = blockIdx.x;
    int n = (int)cnt[b]; if (n > CAP) n = CAP;
    for (int t = threadIdx.x; t < CAP; t += blockDim.x) {
        ls[t] = (t < n) ? cscore[b * CAP + t] : -3e38f;
        li[t] = (t < n) ? cidx[b * CAP + t]   : -1;
    }
    __syncthreads();
    for (int k = 2; k <= CAP; k <<= 1) {
        for (int j = k >> 1; j > 0; j >>= 1) {
            for (int t = threadIdx.x; t < CAP; t += blockDim.x) {
                int ixj = t ^ j;
                if (ixj > t) {
                    bool dirDesc = ((t & k) == 0);
                    float a0 = ls[t], a1 = ls[ixj];
                    bool swap = dirDesc ? (a0 < a1) : (a0 > a1);
                    if (swap) {
                        ls[t] = a1; ls[ixj] = a0;
                        int q = li[t]; li[t] = li[ixj]; li[ixj] = q;
                    }
                }
            }
            __syncthreads();
        }
    }
    for (int i = threadIdx.x; i < 2048; i += blockDim.x) {
        float s = (i < PRE_N) ? ls[i] : -3e38f;
        tscore[b * 2048 + i] = s;
        float bx[4] = {0.f, 0.f, 0.f, 0.f};
        if (i < PRE_N && li[i] >= 0) {
            int j = b * HWA + li[i];
            bx[0] = props[j * 4 + 0]; bx[1] = props[j * 4 + 1];
            bx[2] = props[j * 4 + 2]; bx[3] = props[j * 4 + 3];
        }
        #pragma unroll
        for (int q = 0; q < 4; ++q) tbox[(b * 2048 + i) * 4 + q] = bx[q];
    }
}

// ---------------- K5: greedy NMS over the 2000 sorted boxes ----------------
__global__ __launch_bounds__(256)
void k_nms(const float* __restrict__ tscore, const float* __restrict__ tbox,
           float* __restrict__ mscore) {
    __shared__ float bx[PRE_N * 4];
    __shared__ float ssc[PRE_N];
    __shared__ int   supp[PRE_N];
    int b = blockIdx.x;
    for (int i = threadIdx.x; i < PRE_N; i += blockDim.x) {
        #pragma unroll
        for (int q = 0; q < 4; ++q) bx[i * 4 + q] = tbox[(b * 2048 + i) * 4 + q];
        ssc[i] = tscore[b * 2048 + i];
        supp[i] = 0;
    }
    __syncthreads();
    for (int i = 0; i < PRE_N; ++i) {
        if (!supp[i]) {
            float x1 = bx[i * 4 + 0], y1 = bx[i * 4 + 1];
            float x2 = bx[i * 4 + 2], y2 = bx[i * 4 + 3];
            float ai = (x2 - x1) * (y2 - y1);
            for (int j = i + 1 + threadIdx.x; j < PRE_N; j += blockDim.x) {
                float u1 = fmaxf(x1, bx[j * 4 + 0]);
                float v1 = fmaxf(y1, bx[j * 4 + 1]);
                float u2 = fminf(x2, bx[j * 4 + 2]);
                float v2 = fminf(y2, bx[j * 4 + 3]);
                float iw = fmaxf(u2 - u1, 0.f), ih = fmaxf(v2 - v1, 0.f);
                float inter = iw * ih;
                float aj = (bx[j * 4 + 2] - bx[j * 4 + 0]) * (bx[j * 4 + 3] - bx[j * 4 + 1]);
                float iou = inter / (ai + aj - inter + 1e-12f);
                if (iou > 0.7f) supp[j] = 1;
            }
        }
        __syncthreads();
    }
    for (int i = threadIdx.x; i < PRE_N; i += blockDim.x) {
        bool kept = (!supp[i]) && (ssc[i] > NEGS * 0.5f);
        mscore[b * 2048 + i] = kept ? ssc[i] : -1.0f;
    }
}

// ---------------- K6: stable compaction to 1000, write d_out -------------
__global__ void k_final(const float* __restrict__ mscore, const float* __restrict__ tbox,
                        float* __restrict__ out) {
    int b = blockIdx.x;
    float* out_s = out + b * POST_N;
    float* out_p = out + BB * POST_N + (size_t)b * POST_N * 4;
    for (int i = threadIdx.x; i < POST_N; i += blockDim.x) {
        out_s[i] = -1.0f;
        #pragma unroll
        for (int q = 0; q < 4; ++q) out_p[i * 4 + q] = 0.f;
    }
    __syncthreads();
    if (threadIdx.x == 0) {
        int r = 0;
        for (int i = 0; i < PRE_N && r < POST_N; ++i) {
            float s = mscore[b * 2048 + i];
            if (s > 0.0f) {
                out_s[r] = s;
                #pragma unroll
                for (int q = 0; q < 4; ++q) out_p[r * 4 + q] = tbox[(b * 2048 + i) * 4 + q];
                ++r;
            }
        }
    }
}

// ---------------- host launcher ----------------
extern "C" void kernel_launch(void* const* d_in, const int* in_sizes, int n_in,
                              void* d_out, int out_size, void* d_ws, size_t ws_size,
                              hipStream_t stream) {
    const float* features = (const float*)d_in[0];
    const float* conv_w   = (const float*)d_in[1];
    const float* conv_b   = (const float*)d_in[2];
    const float* cls_w    = (const float*)d_in[3];
    const float* cls_b    = (const float*)d_in[4];
    const float* bbox_w   = (const float*)d_in[5];
    const float* bbox_b   = (const float*)d_in[6];
    const float* anchors  = (const float*)d_in[7];
    (void)in_sizes; (void)n_in; (void)out_size;

    const float fimg_h = 1088.f, fimg_w = 1600.f;

    char* w = (char*)d_ws;
    size_t o = 0;
    auto take = [&](size_t bytes) { char* p = w + o; o = (o + bytes + 255) & ~(size_t)255; return p; };
    unsigned short* wp     = (unsigned short*)take((size_t)9 * CC * CC * 2);
    unsigned short* featp  = (unsigned short*)take((size_t)BB * HP * WP * CC * 2);
    unsigned short* fbuf   = (unsigned short*)take((size_t)NPIX * CC * 2);
    float*          scores = (float*)take((size_t)BB * HWA * 4);
    float*          props  = (float*)take((size_t)BB * HWA * 4 * 4);
    unsigned*       hist   = (unsigned*)take((size_t)BB * 4096 * 4);
    unsigned*       thr    = (unsigned*)take((size_t)BB * 4);
    unsigned*       cnt    = (unsigned*)take((size_t)BB * 4);
    int*            cidx   = (int*)take((size_t)BB * CAP * 4);
    float*          cscore = (float*)take((size_t)BB * CAP * 4);
    float*          tscore = (float*)take((size_t)BB * 2048 * 4);
    float*          tbox   = (float*)take((size_t)BB * 2048 * 4 * 4);
    float*          mscore = (float*)take((size_t)BB * 2048 * 4);
    (void)ws_size;

    hipMemsetAsync(hist, 0, (size_t)BB * 4096 * 4, stream);
    hipMemsetAsync(cnt,  0, (size_t)BB * 4, stream);

    {
        int n = 9 * CC * CC;
        k_pack_w<<<(n + 255) / 256, 256, 0, stream>>>(conv_w, wp);
    }
    {
        int n = BB * HP * WP * CC;
        k_pack_f<<<(n + 255) / 256, 256, 0, stream>>>(features, featp);
    }

    {
        int mtiles = (NPIX + 63) / 64;          // 850
        int ntiles = CC / 128;                  // 2
        k_conv_wmma<<<mtiles * ntiles, 256, 0, stream>>>(featp, wp, conv_b, fbuf);
    }

    k_heads<<<(NPIX + 255) / 256, 256, 0, stream>>>(fbuf, cls_w, cls_b, bbox_w, bbox_b,
                                                    anchors, fimg_w, fimg_h, scores, props);

    {
        int n = BB * HWA;
        k_hist<<<(n + 255) / 256, 256, 0, stream>>>(scores, hist);
        k_thresh<<<BB, 32, 0, stream>>>(hist, thr);
        k_compact<<<(n + 255) / 256, 256, 0, stream>>>(scores, thr, cnt, cidx, cscore);
        k_sort_gather<<<BB, 1024, 0, stream>>>(cnt, cidx, cscore, props, tscore, tbox);
    }

    k_nms<<<BB, 256, 0, stream>>>(tscore, tbox, mscore);
    k_final<<<BB, 256, 0, stream>>>(mscore, tbox, (float*)d_out);
}
